// darwin_64287070487234
// MI455X (gfx1250) — matrix operand order repaired and checked by reading, not compiled, not run on hardware
//
#include <hip/hip_runtime.h>
#include <hip/hip_bf16.h>
#include <stdint.h>

typedef __attribute__((ext_vector_type(16))) _Float16 v16h;
typedef __attribute__((ext_vector_type(8)))  _Float16 h8v;
typedef __attribute__((ext_vector_type(8)))  float    v8f;

#define N_NODES  100000
#define N_EDGES  1600000
#define N_FEAT   92
#define DIM      64
#define N_GRAPHS 64
#define BN_EPS   1e-5f

// -------------------- weight prep: transpose + fp16 convert --------------------
// w1: [3][192][64] -> w1t: [3][64][192] (n-major), w2: [3][64][64] -> [3][64][64] transposed
__global__ void prep_w_kernel(const float* __restrict__ w1,
                              const float* __restrict__ w2,
                              _Float16* __restrict__ w1t,
                              _Float16* __restrict__ w2t) {
  int i = blockIdx.x * blockDim.x + threadIdx.x;
  int stride = gridDim.x * blockDim.x;
  const int n1 = 3 * 192 * 64;
  for (int t = i; t < n1; t += stride) {
    int l = t / (192 * 64);
    int r = t - l * 192 * 64;
    int k = r >> 6;
    int n = r & 63;
    w1t[l * 64 * 192 + n * 192 + k] = (_Float16)w1[t];
  }
  const int n2 = 3 * 64 * 64;
  for (int t = i; t < n2; t += stride) {
    int l = t >> 12;
    int r = t & 4095;
    int k = r >> 6;
    int n = r & 63;
    w2t[l * 64 * 64 + n * 64 + k] = (_Float16)w2[t];
  }
}

// -------------------- pre-linear: h16 = f16(relu(x @ pre_w + pre_b)) --------------------
__global__ void pre_kernel(const float* __restrict__ x,
                           const float* __restrict__ w,
                           const float* __restrict__ b,
                           _Float16* __restrict__ h16) {
  int i = blockIdx.x * blockDim.x + threadIdx.x;
  int stride = gridDim.x * blockDim.x;
  const int tot = N_NODES * DIM;
  for (; i < tot; i += stride) {
    int v = i >> 6, c = i & 63;
    float acc = b[c];
    const float* xr = x + (size_t)v * N_FEAT;
    #pragma unroll 4
    for (int k = 0; k < N_FEAT; ++k) acc = fmaf(xr[k], w[k * DIM + c], acc);
    acc = acc > 0.f ? acc : 0.f;
    h16[i] = (_Float16)acc;
  }
}

// -------------------- edge in-degree counts --------------------
__global__ void count_kernel(const int* __restrict__ dst, float* __restrict__ cnt) {
  int i = blockIdx.x * blockDim.x + threadIdx.x;
  int stride = gridDim.x * blockDim.x;
  for (; i < N_EDGES; i += stride) atomicAdd(&cnt[dst[i]], 1.0f);
}

// -------------------- edge MLP via WMMA: the hot kernel --------------------
// Per wave iteration: 32 edges as two 16-row subtiles sharing every B fragment
// (each LDS B load feeds 2 independent WMMAs).
// m = [x_i | p0*x_j | p1*x_j] (32x192 f16 staged in LDS)
// MLP1: 6 k-steps x 4 n-tiles x 2 subtiles of v_wmma_f32_16x16x32_f16
//       (B streamed from LDS w1 copy); relu; restage 32x64;
// MLP2: 2 k-steps x 4 n-tiles x 2 subtiles (B in registers); relu;
// f32 atomic scatter to agg[dst].
#define MSTR 200         // staging row stride (halfs), multiple of 8 -> 16B rows
#define TSTR 72          // t-tile row stride (halfs)
#define W1_HALFS (64 * 192)                 // 12288 halfs = 24576 B per block
#define WAVE_HALFS (32 * MSTR + 32 * TSTR)  // 8704 halfs = 17408 B per wave
#define NWAVES 2

__global__ __launch_bounds__(32 * NWAVES) void edge_mlp_kernel(
    const _Float16* __restrict__ h16,
    const int* __restrict__ src, const int* __restrict__ dst,
    const float* __restrict__ ea,
    const _Float16* __restrict__ w1t, const _Float16* __restrict__ w2t,
    const float* __restrict__ b1, const float* __restrict__ b2,
    float* __restrict__ agg) {
  // one array: w1 copy [0,12288) then per-wave staging; the shared object makes
  // staging stores alias-unprovable vs w1 loads, keeping B-frags out of registers.
  __shared__ _Float16 smem[W1_HALFS + NWAVES * WAVE_HALFS];
  const int wid = threadIdx.x >> 5;
  const int lane = threadIdx.x & 31;
  const int l15 = lane & 15;
  const int lh = lane >> 4;  // lane half: 0 or 1

  // cooperative copy of this layer's transposed w1 into LDS (1536 uint4)
  {
    const uint4* sw = (const uint4*)w1t;
    uint4* dw = (uint4*)smem;
    for (int i = threadIdx.x; i < W1_HALFS / 8; i += 32 * NWAVES) dw[i] = sw[i];
  }
  __syncthreads();

  const _Float16* lds_w1 = smem;
  _Float16* lds_m = smem + W1_HALFS + wid * WAVE_HALFS;  // 32 x MSTR
  _Float16* lds_t = lds_m + 32 * MSTR;                   // 32 x TSTR

  // w2 B fragments live in registers (8 x v16h = 64 VGPRs)
  v16h w2f[8];
  #pragma unroll
  for (int kt = 0; kt < 2; ++kt) {
    #pragma unroll
    for (int nt = 0; nt < 4; ++nt) {
      const _Float16* bp = w2t + (nt * 16 + l15) * 64 + kt * 32 + lh * 8;
      ((uint4*)&w2f[kt * 4 + nt])[0] = *(const uint4*)bp;
      ((uint4*)&w2f[kt * 4 + nt])[1] = *(const uint4*)(bp + 16);
    }
  }
  float b1v[4], b2v[4];
  #pragma unroll
  for (int nt = 0; nt < 4; ++nt) {
    b1v[nt] = b1[nt * 16 + l15];
    b2v[nt] = b2[nt * 16 + l15];
  }

  const int nSuper = N_EDGES / 32;
  for (int st = blockIdx.x * NWAVES + wid; st < nSuper; st += gridDim.x * NWAVES) {
    const int e0 = st * 32;

    // ---- stage message tile [32 x 192] into LDS (lane: 2 edges, cols lh*32..) ----
    #pragma unroll
    for (int t2 = 0; t2 < 2; ++t2) {
      const int e = e0 + t2 * 16 + l15;
      const int s = src[e], d = dst[e];
      const float2 p = *(const float2*)(ea + 2 * e);
      const _Float16 p0h = (_Float16)p.x, p1h = (_Float16)p.y;
      const _Float16* hd = h16 + (size_t)d * DIM + lh * 32;
      const _Float16* hs = h16 + (size_t)s * DIM + lh * 32;
      _Float16* row = lds_m + (t2 * 16 + l15) * MSTR;
      const int cb = lh * 32;
      #pragma unroll
      for (int c = 0; c < 32; c += 8) {
        h8v vi = *(const h8v*)(hd + c);
        *(h8v*)(row + cb + c) = vi;              // x_i
        h8v vj = *(const h8v*)(hs + c);
        *(h8v*)(row + 64 + cb + c) = vj * p0h;   // v_pk_mul_f16
        *(h8v*)(row + 128 + cb + c) = vj * p1h;
      }
    }

    // ---- MLP1: [32,192] @ [192,64]; each B fragment feeds 2 WMMAs ----
    v8f acc[2][4];
    #pragma unroll
    for (int t2 = 0; t2 < 2; ++t2) {
      #pragma unroll
      for (int nt = 0; nt < 4; ++nt) {
        #pragma unroll
        for (int r = 0; r < 8; ++r) acc[t2][nt][r] = b1v[nt];
      }
    }
    #pragma unroll
    for (int kt = 0; kt < 6; ++kt) {
      v16h a0, a1;  // A frags for the two subtiles
      const _Float16* ap0 = lds_m + l15 * MSTR + kt * 32 + lh * 8;
      const _Float16* ap1 = ap0 + 16 * MSTR;
      ((uint4*)&a0)[0] = *(const uint4*)ap0;
      ((uint4*)&a0)[1] = *(const uint4*)(ap0 + 16);
      ((uint4*)&a1)[0] = *(const uint4*)ap1;
      ((uint4*)&a1)[1] = *(const uint4*)(ap1 + 16);
      #pragma unroll
      for (int nt = 0; nt < 4; ++nt) {
        v16h bfr;  // B frag from LDS w1 copy: col=l15, K = lh*8 + i (i<8), 16 + lh*8 + (i-8) (i>=8)
        const _Float16* bp = lds_w1 + (nt * 16 + l15) * 192 + kt * 32 + lh * 8;
        ((uint4*)&bfr)[0] = *(const uint4*)bp;
        ((uint4*)&bfr)[1] = *(const uint4*)(bp + 16);
        acc[0][nt] = __builtin_amdgcn_wmma_f32_16x16x32_f16(
            false, a0, false, bfr, (short)0, acc[0][nt], false, false);
        acc[1][nt] = __builtin_amdgcn_wmma_f32_16x16x32_f16(
            false, a1, false, bfr, (short)0, acc[1][nt], false, false);
      }
    }

    // ---- relu + restage t tile [32 x 64] ----
    #pragma unroll
    for (int t2 = 0; t2 < 2; ++t2) {
      #pragma unroll
      for (int nt = 0; nt < 4; ++nt) {
        #pragma unroll
        for (int r = 0; r < 8; ++r) {
          float v = acc[t2][nt][r];
          v = v > 0.f ? v : 0.f;
          lds_t[(t2 * 16 + r + 8 * lh) * TSTR + nt * 16 + l15] = (_Float16)v;
        }
      }
    }

    // ---- MLP2: [32,64] @ [64,64], B in registers ----
    v8f o[2][4];
    #pragma unroll
    for (int t2 = 0; t2 < 2; ++t2) {
      #pragma unroll
      for (int nt = 0; nt < 4; ++nt) {
        #pragma unroll
        for (int r = 0; r < 8; ++r) o[t2][nt][r] = b2v[nt];
      }
    }
    #pragma unroll
    for (int kt = 0; kt < 2; ++kt) {
      v16h a0, a1;
      const _Float16* ap0 = lds_t + l15 * TSTR + kt * 32 + lh * 8;
      const _Float16* ap1 = ap0 + 16 * TSTR;
      ((uint4*)&a0)[0] = *(const uint4*)ap0;
      ((uint4*)&a0)[1] = *(const uint4*)(ap0 + 16);
      ((uint4*)&a1)[0] = *(const uint4*)ap1;
      ((uint4*)&a1)[1] = *(const uint4*)(ap1 + 16);
      #pragma unroll
      for (int nt = 0; nt < 4; ++nt) {
        o[0][nt] = __builtin_amdgcn_wmma_f32_16x16x32_f16(
            false, a0, false, w2f[kt * 4 + nt], (short)0, o[0][nt], false, false);
        o[1][nt] = __builtin_amdgcn_wmma_f32_16x16x32_f16(
            false, a1, false, w2f[kt * 4 + nt], (short)0, o[1][nt], false, false);
      }
    }

    // ---- relu + scatter (f32 global atomics, L2-resident agg) ----
    #pragma unroll
    for (int t2 = 0; t2 < 2; ++t2) {
      int dl[8];
      #pragma unroll
      for (int r = 0; r < 8; ++r) dl[r] = dst[e0 + t2 * 16 + r + 8 * lh];
      #pragma unroll
      for (int nt = 0; nt < 4; ++nt) {
        #pragma unroll
        for (int r = 0; r < 8; ++r) {
          float v = o[t2][nt][r];
          v = v > 0.f ? v : 0.f;
          atomicAdd(&agg[(size_t)dl[r] * DIM + nt * 16 + l15], v);
        }
      }
    }
  }
}

// -------------------- column stats (sum, sumsq of agg/cnt) --------------------
__global__ __launch_bounds__(256) void stats_kernel(const float* __restrict__ agg,
                                                    const float* __restrict__ cnt,
                                                    float* __restrict__ colstats) {
  __shared__ float sh[2][4][64];
  const int c = threadIdx.x & 63;
  const int rl = threadIdx.x >> 6;  // 0..3
  float s = 0.f, ss = 0.f;
  for (int v = blockIdx.x * 4 + rl; v < N_NODES; v += gridDim.x * 4) {
    float cn = cnt[v];
    cn = cn > 1.f ? cn : 1.f;
    float hv = agg[(size_t)v * DIM + c] / cn;
    s += hv;
    ss += hv * hv;
  }
  sh[0][rl][c] = s;
  sh[1][rl][c] = ss;
  __syncthreads();
  if (threadIdx.x < 64) {
    float ts = 0.f, tss = 0.f;
    #pragma unroll
    for (int r = 0; r < 4; ++r) { ts += sh[0][r][c]; tss += sh[1][r][c]; }
    atomicAdd(&colstats[c], ts);
    atomicAdd(&colstats[64 + c], tss);
  }
}

// -------------------- BN finalize -> h16 --------------------
__global__ void bn_kernel(const float* __restrict__ agg,
                          const float* __restrict__ cnt,
                          const float* __restrict__ colstats,
                          const float* __restrict__ g,
                          const float* __restrict__ b,
                          _Float16* __restrict__ h16) {
  int i = blockIdx.x * blockDim.x + threadIdx.x;
  int stride = gridDim.x * blockDim.x;
  const int tot = N_NODES * DIM;
  const float invN = 1.0f / (float)N_NODES;
  for (; i < tot; i += stride) {
    int v = i >> 6, c = i & 63;
    float mu = colstats[c] * invN;
    float var = colstats[64 + c] * invN - mu * mu;
    float rs = rsqrtf(var + BN_EPS);
    float cn = cnt[v];
    cn = cn > 1.f ? cn : 1.f;
    float hv = agg[i] / cn;
    h16[i] = (_Float16)((hv - mu) * rs * g[c] + b[c]);
  }
}

// -------------------- global mean pool --------------------
__global__ void pool_kernel(const _Float16* __restrict__ h16,
                            const int* __restrict__ batch,
                            float* __restrict__ pool,
                            float* __restrict__ pcnt) {
  int i = blockIdx.x * blockDim.x + threadIdx.x;
  int stride = gridDim.x * blockDim.x;
  const int tot = N_NODES * DIM;
  for (; i < tot; i += stride) {
    int v = i >> 6, c = i & 63;
    int gidx = batch[v];
    atomicAdd(&pool[gidx * DIM + c], (float)h16[i]);
    if (c == 0) atomicAdd(&pcnt[gidx], 1.0f);
  }
}

// -------------------- head: relu(pool @ post_w + post_b) @ out_w + out_b --------------------
__global__ __launch_bounds__(64) void head_kernel(const float* __restrict__ pool,
                                                  const float* __restrict__ pcnt,
                                                  const float* __restrict__ post_w,
                                                  const float* __restrict__ post_b,
                                                  const float* __restrict__ out_w,
                                                  const float* __restrict__ out_b,
                                                  float* __restrict__ out) {
  int g = threadIdx.x;
  if (g >= N_GRAPHS) return;
  float p[DIM];
  float cn = pcnt[g];
  cn = cn > 1.f ? cn : 1.f;
  #pragma unroll 8
  for (int c = 0; c < DIM; ++c) p[c] = pool[g * DIM + c] / cn;
  float accum = out_b[0];
  for (int j = 0; j < DIM; ++j) {
    float y = post_b[j];
    #pragma unroll 8
    for (int c = 0; c < DIM; ++c) y = fmaf(p[c], post_w[c * DIM + j], y);
    y = y > 0.f ? y : 0.f;
    accum = fmaf(y, out_w[j], accum);
  }
  out[g] = accum;
}

// -------------------- launch --------------------
extern "C" void kernel_launch(void* const* d_in, const int* in_sizes, int n_in,
                              void* d_out, int out_size, void* d_ws, size_t ws_size,
                              hipStream_t stream) {
  const float* x       = (const float*)d_in[0];
  const float* ea      = (const float*)d_in[1];
  const float* pre_w   = (const float*)d_in[2];
  const float* pre_b   = (const float*)d_in[3];
  const float* conv_w1 = (const float*)d_in[4];
  const float* conv_b1 = (const float*)d_in[5];
  const float* conv_w2 = (const float*)d_in[6];
  const float* conv_b2 = (const float*)d_in[7];
  const float* bn_g    = (const float*)d_in[8];
  const float* bn_b    = (const float*)d_in[9];
  const float* post_w  = (const float*)d_in[10];
  const float* post_b  = (const float*)d_in[11];
  const float* out_w   = (const float*)d_in[12];
  const float* out_b   = (const float*)d_in[13];
  const int*   eidx    = (const int*)d_in[14];
  const int*   batch   = (const int*)d_in[15];
  float* out = (float*)d_out;

  const int* src = eidx;
  const int* dst = eidx + N_EDGES;

  // workspace carve-out (256B aligned)
  char* ws = (char*)d_ws;
  size_t off = 0;
  auto carve = [&](size_t bytes) -> char* {
    char* p = ws + off;
    off = (off + bytes + 255) & ~(size_t)255;
    return p;
  };
  _Float16* h16      = (_Float16*)carve((size_t)N_NODES * DIM * 2);
  float*    agg      = (float*)   carve((size_t)N_NODES * DIM * 4);
  float*    cnt      = (float*)   carve((size_t)N_NODES * 4);
  float*    colstats = (float*)   carve(128 * 4);
  _Float16* w1t      = (_Float16*)carve((size_t)3 * 64 * 192 * 2);
  _Float16* w2t      = (_Float16*)carve((size_t)3 * 64 * 64 * 2);
  float*    pool     = (float*)   carve((size_t)N_GRAPHS * DIM * 4);
  float*    pcnt     = (float*)   carve((size_t)N_GRAPHS * 4);
  (void)ws_size;

  hipMemsetAsync(cnt, 0, (size_t)N_NODES * 4, stream);
  hipMemsetAsync(pool, 0, (size_t)N_GRAPHS * DIM * 4, stream);
  hipMemsetAsync(pcnt, 0, (size_t)N_GRAPHS * 4, stream);

  prep_w_kernel<<<64, 256, 0, stream>>>(conv_w1, conv_w2, w1t, w2t);
  pre_kernel<<<8192, 256, 0, stream>>>(x, pre_w, pre_b, h16);
  count_kernel<<<2048, 256, 0, stream>>>(dst, cnt);

  for (int l = 0; l < 3; ++l) {
    hipMemsetAsync(agg, 0, (size_t)N_NODES * DIM * 4, stream);
    hipMemsetAsync(colstats, 0, 128 * 4, stream);
    edge_mlp_kernel<<<2048, 32 * NWAVES, 0, stream>>>(
        h16, src, dst, ea,
        w1t + (size_t)l * 64 * 192, w2t + (size_t)l * 64 * 64,
        conv_b1 + l * DIM, conv_b2 + l * DIM, agg);
    stats_kernel<<<512, 256, 0, stream>>>(agg, cnt, colstats);
    bn_kernel<<<8192, 256, 0, stream>>>(agg, cnt, colstats,
                                        bn_g + l * DIM, bn_b + l * DIM, h16);
  }

  pool_kernel<<<8192, 256, 0, stream>>>(h16, batch, pool, pcnt);
  head_kernel<<<1, 64, 0, stream>>>(pool, pcnt, post_w, post_b, out_w, out_b, out);
}
